// DoubleStreamBlockIPA_283467841988
// MI455X (gfx1250) — matrix operand order repaired and checked by reading, not compiled, not run on hardware
//
#include <hip/hip_runtime.h>
#include <math.h>

// ---------------------------------------------------------------------------
// DoubleStreamBlock w/ IP-adapter for MI455X (gfx1250, wave32, WMMA).
// All heavy GEMMs run on v_wmma_f32_16x16x32_bf16 with f32 accumulation.
// bf16 is the storage format for every matrix operand: weights are converted
// once per call, producer kernels emit bf16, so the GEMM hot loop is pure
// load -> LDS -> wmma with no conversion VALU.
// Block tile 128x128, wave tile 64x32: 8 wmma per wave k-step from 12
// ds_load_b128 (reuse 2.67), software-pipelined global->reg->LDS staging.
// ---------------------------------------------------------------------------

#define HDIM   2048
#define NHEADS 16
#define DHEAD  128
#define L_IMG  1024
#define L_TXT  256
#define L_SEQ  1280
#define MLPD   8192
#define N_IP   16
#define EPSV   1e-6f
#define ATT_SCALE 0.08838834764831845f   // 1/sqrt(128)

typedef __attribute__((ext_vector_type(16))) __bf16 v16bf;
typedef __attribute__((ext_vector_type(8)))  float  v8f;
typedef unsigned short u16;

// epilogue flags
#define EP_BIAS    1
#define EP_GELU    2
#define EP_RESGATE 4

__device__ __forceinline__ u16 f2bf(float f) {
  unsigned int u = __float_as_uint(f);
  u += 0x7fffu + ((u >> 16) & 1u);
  return (u16)(u >> 16);
}

union Frag16 { u16 u[16]; v16bf v; };

// A fragment: 16x32 bf16.  lane<16: K{0..7,16..23}; lane>=16: K{8..15,24..31}
__device__ __forceinline__ v16bf ldfragA(const u16 (*tile)[40], int rbase, int lane) {
  const u16* p = tile[rbase + (lane & 15)];
  const int kb = (lane >> 4) ? 8 : 0;
  Frag16 r;
#pragma unroll
  for (int i = 0; i < 8; ++i) { r.u[i] = p[kb + i]; r.u[8 + i] = p[kb + 16 + i]; }
  return r.v;
}

// B fragment: 32x16 bf16 stored in LDS as [n][k]; lanes 0-15: K{0..7,16..23},
// lanes 16-31: K{8..15,24..31}, column N = lane%16.
__device__ __forceinline__ v16bf ldfragB(const u16 (*tile)[40], int nbase, int lane) {
  const u16* p = tile[nbase + (lane & 15)];
  const int kb = (lane >> 4) ? 8 : 0;
  Frag16 r;
#pragma unroll
  for (int i = 0; i < 8; ++i) { r.u[i] = p[kb + i]; r.u[8 + i] = p[kb + 16 + i]; }
  return r.v;
}

// ---------------------------------------------------------------------------
// WMMA GEMM, bf16 in / f32 accum:  C = epi(alpha * A(MxK) @ B(KxN))
//   TRANSB: B element (k,n) = Bsrc[n*ldb + k]   (k-major source)
//   GUARD:  M not a multiple of 128 (row guards)
//   OUTBF16: store C as bf16 (C16) instead of f32 (C)
// Block tile 128x128, BK=32, 8 waves -> 64x32/wave -> 8 wmma per k-step.
// ---------------------------------------------------------------------------
template<int TRANSB, int GUARD, int OUTBF16>
__global__ void __launch_bounds__(256)
gemm_bf16_kernel(const u16* __restrict__ A, long lda,
                 const u16* __restrict__ B, long ldb,
                 float* __restrict__ C, u16* __restrict__ C16, long ldc,
                 const float* __restrict__ bias,
                 const float* __restrict__ gate,
                 const float* __restrict__ res, long ldres,
                 int M, int N, int K, float alpha, int flags) {
  __shared__ u16 As[128][40];
  __shared__ u16 Bs[128][40];

  const int tid  = threadIdx.x;
  const int lane = tid & 31;
  const int wave = tid >> 5;
  const int wm = (wave >> 2) * 64;   // 0 / 64
  const int wn = (wave & 3) * 32;    // 0..96
  const int rowBase = blockIdx.y * 128;
  const int colBase = blockIdx.x * 128;

  const int aR = tid >> 3;            // 0..31 (chunk i adds 32)
  const int aC = (tid & 7) << 2;      // 0..28

  uint2 aReg[4], bReg[4];

  auto loadTiles = [&](int k0) {
#pragma unroll
    for (int i = 0; i < 4; ++i) {
      int gr = rowBase + aR + i * 32;
      if (GUARD && gr >= M) { aReg[i].x = 0u; aReg[i].y = 0u; }
      else aReg[i] = *(const uint2*)(A + (long)gr * lda + k0 + aC);
    }
    if (TRANSB) {
#pragma unroll
      for (int i = 0; i < 4; ++i) {
        int idx = tid + i * 256;
        int n  = idx >> 3;              // 0..127
        int kk = (idx & 7) << 2;        // 0..28
        bReg[i] = *(const uint2*)(B + (long)(colBase + n) * ldb + k0 + kk);
      }
    } else {
#pragma unroll
      for (int i = 0; i < 4; ++i) {
        int idx = tid + i * 256;
        int kk = idx >> 5;              // 0..31
        int n  = (idx & 31) << 2;       // 0..124
        bReg[i] = *(const uint2*)(B + (long)(k0 + kk) * ldb + colBase + n);
      }
    }
  };

  auto storeTiles = [&]() {
#pragma unroll
    for (int i = 0; i < 4; ++i)
      *(uint2*)&As[aR + i * 32][aC] = aReg[i];
    if (TRANSB) {
#pragma unroll
      for (int i = 0; i < 4; ++i) {
        int idx = tid + i * 256;
        int n  = idx >> 3;
        int kk = (idx & 7) << 2;
        *(uint2*)&Bs[n][kk] = bReg[i];
      }
    } else {
#pragma unroll
      for (int i = 0; i < 4; ++i) {
        int idx = tid + i * 256;
        int kk = idx >> 5;
        int n  = (idx & 31) << 2;
        const u16* pr = (const u16*)&bReg[i];
        Bs[n + 0][kk] = pr[0]; Bs[n + 1][kk] = pr[1];
        Bs[n + 2][kk] = pr[2]; Bs[n + 3][kk] = pr[3];
      }
    }
  };

  v8f acc[4][2];
#pragma unroll
  for (int i = 0; i < 4; ++i)
#pragma unroll
    for (int j = 0; j < 2; ++j)
#pragma unroll
      for (int r = 0; r < 8; ++r) acc[i][j][r] = 0.0f;

  loadTiles(0);
  for (int k0 = 0; k0 < K; k0 += 32) {
    storeTiles();
    __syncthreads();
    if (k0 + 32 < K) loadTiles(k0 + 32);   // prefetch next tile into regs

    v16bf b0 = ldfragB(Bs, wn,      lane);
    v16bf b1 = ldfragB(Bs, wn + 16, lane);
#pragma unroll
    for (int ti = 0; ti < 4; ++ti) {
      v16bf a = ldfragA(As, wm + ti * 16, lane);
      acc[ti][0] = __builtin_amdgcn_wmma_f32_16x16x32_bf16(false, a, false, b0, (short)0, acc[ti][0], false, false);
      acc[ti][1] = __builtin_amdgcn_wmma_f32_16x16x32_bf16(false, a, false, b1, (short)0, acc[ti][1], false, false);
    }
    __syncthreads();
  }

  // epilogue.  C layout: VGPR r -> M = r + 8*(lane>=16), N = lane&15
  const int nloc  = lane & 15;
  const int mhalf = (lane >> 4) * 8;
#pragma unroll
  for (int ti = 0; ti < 4; ++ti) {
#pragma unroll
    for (int tj = 0; tj < 2; ++tj) {
      int col = colBase + wn + tj * 16 + nloc;
#pragma unroll
      for (int r = 0; r < 8; ++r) {
        int row = rowBase + wm + ti * 16 + mhalf + r;
        if (GUARD && row >= M) continue;
        float c = acc[ti][tj][r] * alpha;
        if (flags & EP_BIAS) c += bias[col];
        if (flags & EP_GELU) {
          float x = c;
          float inner = 0.7978845608028654f * (x + 0.044715f * x * x * x);
          c = 0.5f * x * (1.0f + tanhf(inner));
        }
        if (flags & EP_RESGATE) c = res[(long)row * ldres + col] + gate[col] * c;
        if (OUTBF16) C16[(long)row * ldc + col] = f2bf(c);
        else         C[(long)row * ldc + col] = c;
      }
    }
  }
}

// ---------------------------------------------------------------------------
// bulk f32 -> bf16 conversion (8 elements / thread; n multiple of 8)
// ---------------------------------------------------------------------------
__global__ void __launch_bounds__(256)
cvt_bf16_kernel(const float* __restrict__ x, u16* __restrict__ y, long n) {
  long i = ((long)blockIdx.x * 256 + threadIdx.x) * 8;
  if (i >= n) return;
  float4 a = *(const float4*)(x + i);
  float4 b = *(const float4*)(x + i + 4);
  uint4 pk;
  pk.x = (unsigned)f2bf(a.x) | ((unsigned)f2bf(a.y) << 16);
  pk.y = (unsigned)f2bf(a.z) | ((unsigned)f2bf(a.w) << 16);
  pk.z = (unsigned)f2bf(b.x) | ((unsigned)f2bf(b.y) << 16);
  pk.w = (unsigned)f2bf(b.z) | ((unsigned)f2bf(b.w) << 16);
  *(uint4*)(y + i) = pk;
}

// ---------------------------------------------------------------------------
// silu(vec) -> sv
// ---------------------------------------------------------------------------
__global__ void silu_kernel(const float* __restrict__ vec, float* __restrict__ sv) {
  int i = blockIdx.x * 256 + threadIdx.x;
  if (i < HDIM) { float x = vec[i]; sv[i] = x / (1.0f + expf(-x)); }
}

// m[j] = sum_i sv[i]*w[i,j] + b[j]
__global__ void __launch_bounds__(256)
mod_kernel(const float* __restrict__ sv, const float* __restrict__ w,
           const float* __restrict__ b, float* __restrict__ m) {
  int j = blockIdx.x * 256 + threadIdx.x;
  float acc = 0.f;
  for (int i = 0; i < HDIM; ++i) acc += sv[i] * w[(long)i * (6 * HDIM) + j];
  m[j] = acc + b[j];
}

// ---------------------------------------------------------------------------
// y_bf16 = (1+sc[c]) * LN(x)[row,c] + sh[c]   (one block per row)
// ---------------------------------------------------------------------------
__global__ void __launch_bounds__(256)
lnmod_kernel(const float* __restrict__ x, const float* __restrict__ sh,
             const float* __restrict__ sc, u16* __restrict__ y) {
  __shared__ float r1[256], r2[256];
  const int row = blockIdx.x;
  const int tid = threadIdx.x;
  const float* xr = x + (long)row * HDIM;
  float s = 0.f, s2 = 0.f;
  for (int c = tid; c < HDIM; c += 256) { float v = xr[c]; s += v; s2 += v * v; }
  r1[tid] = s; r2[tid] = s2;
  __syncthreads();
  for (int o = 128; o > 0; o >>= 1) {
    if (tid < o) { r1[tid] += r1[tid + o]; r2[tid] += r2[tid + o]; }
    __syncthreads();
  }
  float mu  = r1[0] * (1.0f / HDIM);
  float var = r2[0] * (1.0f / HDIM) - mu * mu;
  float rstd = rsqrtf(var + EPSV);
  u16* yr = y + (long)row * HDIM;
  for (int c = tid; c < HDIM; c += 256)
    yr[c] = f2bf((1.0f + sc[c]) * ((xr[c] - mu) * rstd) + sh[c]);
}

// ---------------------------------------------------------------------------
// QKV finish: per-head RMS(q,k), RoPE at concat position, repack to
// head-major bf16 [h][l][d].  Keeps pre-RoPE img q (f32) for IP attention.
// grid = (nrows, NHEADS), block = 128
// ---------------------------------------------------------------------------
__global__ void __launch_bounds__(128)
qkvfin_kernel(const float* __restrict__ qkv, int nrows, int seqoff,
              const float* __restrict__ qscale, const float* __restrict__ kscale,
              const float* __restrict__ pe,
              u16* __restrict__ qpk, u16* __restrict__ kpk,
              u16* __restrict__ vpk, float* __restrict__ qnope) {
  __shared__ float red[128];
  __shared__ float qs[128], ks[128];
  const int i = blockIdx.x;
  const int h = blockIdx.y;
  const int d = threadIdx.x;
  const int l = seqoff + i;
  const float* row = qkv + (long)i * (3 * HDIM);
  float q = row[h * DHEAD + d];
  float k = row[HDIM + h * DHEAD + d];
  float v = row[2 * HDIM + h * DHEAD + d];

  red[d] = q * q; __syncthreads();
  for (int o = 64; o > 0; o >>= 1) { if (d < o) red[d] += red[d + o]; __syncthreads(); }
  float qn = q * rsqrtf(red[0] * (1.0f / DHEAD) + EPSV) * qscale[d];
  __syncthreads();
  red[d] = k * k; __syncthreads();
  for (int o = 64; o > 0; o >>= 1) { if (d < o) red[d] += red[d + o]; __syncthreads(); }
  float kn = k * rsqrtf(red[0] * (1.0f / DHEAD) + EPSV) * kscale[d];
  __syncthreads();

  qs[d] = qn; ks[d] = kn;
  __syncthreads();

  int j = d >> 1, t = d & 1;
  long peb = ((long)l * 64 + j) * 4 + t * 2;
  float p0 = pe[peb], p1 = pe[peb + 1];
  float qr = p0 * qs[2 * j] + p1 * qs[2 * j + 1];
  float kr = p0 * ks[2 * j] + p1 * ks[2 * j + 1];

  long oidx = ((long)h * L_SEQ + l) * DHEAD + d;
  qpk[oidx] = f2bf(qr); kpk[oidx] = f2bf(kr); vpk[oidx] = f2bf(v);
  if (qnope) qnope[((long)h * nrows + i) * DHEAD + d] = qn;
}

// ---------------------------------------------------------------------------
// Row softmax: S (f32, rows x cols) -> P (bf16, normalized)
// ---------------------------------------------------------------------------
__global__ void __launch_bounds__(256)
softmax_kernel(float* __restrict__ S, u16* __restrict__ P, int cols) {
  __shared__ float red[256];
  const int row = blockIdx.x;
  const int tid = threadIdx.x;
  float* sr = S + (long)row * cols;
  u16*   pr = P + (long)row * cols;
  float mx = -1e30f;
  for (int c = tid; c < cols; c += 256) mx = fmaxf(mx, sr[c]);
  red[tid] = mx; __syncthreads();
  for (int o = 128; o > 0; o >>= 1) { if (tid < o) red[tid] = fmaxf(red[tid], red[tid + o]); __syncthreads(); }
  mx = red[0]; __syncthreads();
  float sum = 0.f;
  for (int c = tid; c < cols; c += 256) { float e = expf(sr[c] - mx); sr[c] = e; sum += e; }
  red[tid] = sum; __syncthreads();
  for (int o = 128; o > 0; o >>= 1) { if (tid < o) red[tid] += red[tid + o]; __syncthreads(); }
  float inv = 1.0f / red[0];
  for (int c = tid; c < cols; c += 256) pr[c] = f2bf(sr[c] * inv);
}

// ---------------------------------------------------------------------------
// IP attention (16 keys): attn[img rows] += softmax(qnope@ipk^T*s)@ipv
// ---------------------------------------------------------------------------
__global__ void __launch_bounds__(128)
ipa_kernel(const float* __restrict__ qnope, const float* __restrict__ ipk,
           const float* __restrict__ ipv, float* __restrict__ attn) {
  __shared__ float red[128];
  __shared__ float s[N_IP];
  const int i = blockIdx.x;
  const int h = blockIdx.y;
  const int d = threadIdx.x;
  float q = qnope[((long)h * L_IMG + i) * DHEAD + d];
  for (int n = 0; n < N_IP; ++n) {
    red[d] = q * ipk[(long)n * HDIM + h * DHEAD + d];
    __syncthreads();
    for (int o = 64; o > 0; o >>= 1) { if (d < o) red[d] += red[d + o]; __syncthreads(); }
    if (d == 0) s[n] = red[0] * ATT_SCALE;
    __syncthreads();
  }
  float mx = -1e30f;
  for (int n = 0; n < N_IP; ++n) mx = fmaxf(mx, s[n]);
  float p[N_IP], sum = 0.f;
  for (int n = 0; n < N_IP; ++n) { p[n] = expf(s[n] - mx); sum += p[n]; }
  float o = 0.f;
  for (int n = 0; n < N_IP; ++n) o += p[n] * ipv[(long)n * HDIM + h * DHEAD + d];
  attn[(long)(L_TXT + i) * HDIM + h * DHEAD + d] += o / sum;   // IP_SCALE = 1
}

// ---------------------------------------------------------------------------
// host side
// ---------------------------------------------------------------------------
static void launch_gemm(hipStream_t st, const u16* A, long lda,
                        const u16* B, long ldb, int transB,
                        float* C, u16* C16, long ldc,
                        const float* bias, const float* gate,
                        const float* res, long ldres,
                        int M, int N, int K, float alpha, int flags) {
  dim3 g(N / 128, (M + 127) / 128), b(256);
  const bool guard = (M & 127) != 0;
  if (transB)
    gemm_bf16_kernel<1, 0, 0><<<g, b, 0, st>>>(A, lda, B, ldb, C, C16, ldc, bias, gate, res, ldres, M, N, K, alpha, flags);
  else if (guard)
    gemm_bf16_kernel<0, 1, 0><<<g, b, 0, st>>>(A, lda, B, ldb, C, C16, ldc, bias, gate, res, ldres, M, N, K, alpha, flags);
  else if (C16)
    gemm_bf16_kernel<0, 0, 1><<<g, b, 0, st>>>(A, lda, B, ldb, C, C16, ldc, bias, gate, res, ldres, M, N, K, alpha, flags);
  else
    gemm_bf16_kernel<0, 0, 0><<<g, b, 0, st>>>(A, lda, B, ldb, C, C16, ldc, bias, gate, res, ldres, M, N, K, alpha, flags);
}

extern "C" void kernel_launch(void* const* d_in, const int* in_sizes, int n_in,
                              void* d_out, int out_size, void* d_ws, size_t ws_size,
                              hipStream_t stream) {
  const float* img        = (const float*)d_in[0];
  const float* txt        = (const float*)d_in[1];
  const float* vec        = (const float*)d_in[2];
  const float* pe         = (const float*)d_in[3];
  const float* image_emb  = (const float*)d_in[5];
  const float* img_mod_w  = (const float*)d_in[6];
  const float* img_mod_b  = (const float*)d_in[7];
  const float* txt_mod_w  = (const float*)d_in[8];
  const float* txt_mod_b  = (const float*)d_in[9];
  const float* img_qkv_w  = (const float*)d_in[10];
  const float* img_qkv_b  = (const float*)d_in[11];
  const float* txt_qkv_w  = (const float*)d_in[12];
  const float* txt_qkv_b  = (const float*)d_in[13];
  const float* img_q_scale = (const float*)d_in[14];
  const float* img_k_scale = (const float*)d_in[15];
  const float* txt_q_scale = (const float*)d_in[16];
  const float* txt_k_scale = (const float*)d_in[17];
  const float* img_proj_w = (const float*)d_in[18];
  const float* img_proj_b = (const float*)d_in[19];
  const float* txt_proj_w = (const float*)d_in[20];
  const float* txt_proj_b = (const float*)d_in[21];
  const float* img_mlp_w1 = (const float*)d_in[22];
  const float* img_mlp_b1 = (const float*)d_in[23];
  const float* img_mlp_w2 = (const float*)d_in[24];
  const float* img_mlp_b2 = (const float*)d_in[25];
  const float* txt_mlp_w1 = (const float*)d_in[26];
  const float* txt_mlp_b1 = (const float*)d_in[27];
  const float* txt_mlp_w2 = (const float*)d_in[28];
  const float* txt_mlp_b2 = (const float*)d_in[29];
  const float* ip_k_w     = (const float*)d_in[30];
  const float* ip_k_b     = (const float*)d_in[31];
  const float* ip_v_w     = (const float*)d_in[32];
  const float* ip_v_b     = (const float*)d_in[33];
  (void)in_sizes; (void)n_in; (void)out_size; (void)ws_size;

  float* out_img = (float*)d_out;                        // 1024 x 2048
  float* out_txt = (float*)d_out + (size_t)L_IMG * HDIM; // 256 x 2048

  // --- workspace layout (byte cursor, 256B aligned regions) ---
  char* base = (char*)d_ws;
  size_t off = 0;
  auto allocB = [&](size_t bytes) -> char* {
    char* p = base + off; off += (bytes + 255) & ~(size_t)255; return p;
  };
  auto allocF = [&](size_t n) { return (float*)allocB(n * 4); };
  auto allocH = [&](size_t n) { return (u16*)allocB(n * 2); };

  // bf16 weights
  u16* wqkv_i  = allocH((size_t)HDIM * 3 * HDIM);
  u16* wqkv_t  = allocH((size_t)HDIM * 3 * HDIM);
  u16* wproj_i = allocH((size_t)HDIM * HDIM);
  u16* wproj_t = allocH((size_t)HDIM * HDIM);
  u16* wmlp1_i = allocH((size_t)HDIM * MLPD);
  u16* wmlp1_t = allocH((size_t)HDIM * MLPD);
  u16* wmlp2_i = allocH((size_t)MLPD * HDIM);
  u16* wmlp2_t = allocH((size_t)MLPD * HDIM);
  u16* wipk    = allocH((size_t)HDIM * HDIM);
  u16* wipv    = allocH((size_t)HDIM * HDIM);
  u16* emb16   = allocH((size_t)N_IP * HDIM);
  // activations
  float* sv      = allocF(HDIM);
  float* mod_img = allocF(6 * HDIM);
  float* mod_txt = allocF(6 * HDIM);
  u16*   xmod_i  = allocH((size_t)L_IMG * HDIM);
  u16*   xmod_t  = allocH((size_t)L_TXT * HDIM);
  float* qkv_i   = allocF((size_t)L_IMG * 3 * HDIM);
  float* qkv_t   = allocF((size_t)L_TXT * 3 * HDIM);
  u16*   qpk     = allocH((size_t)NHEADS * L_SEQ * DHEAD);
  u16*   kpk     = allocH((size_t)NHEADS * L_SEQ * DHEAD);
  u16*   vpk     = allocH((size_t)NHEADS * L_SEQ * DHEAD);
  float* qnope   = allocF((size_t)NHEADS * L_IMG * DHEAD);
  float* ipk     = allocF((size_t)N_IP * HDIM);
  float* ipv     = allocF((size_t)N_IP * HDIM);
  float* Sbuf    = allocF((size_t)L_SEQ * L_SEQ);
  u16*   Pbuf    = allocH((size_t)L_SEQ * L_SEQ);
  float* attn    = allocF((size_t)L_SEQ * HDIM);
  u16*   attn16  = allocH((size_t)L_SEQ * HDIM);
  float* mid_i   = allocF((size_t)L_IMG * HDIM);
  float* mid_t   = allocF((size_t)L_TXT * HDIM);
  u16*   x2_i    = allocH((size_t)L_IMG * HDIM);
  u16*   x2_t    = allocH((size_t)L_TXT * HDIM);
  u16*   hid_i   = allocH((size_t)L_IMG * MLPD);
  u16*   hid_t   = allocH((size_t)L_TXT * MLPD);

  auto cvt = [&](const float* src, u16* dst, long n) {
    long blocks = (n + 2047) / 2048;
    cvt_bf16_kernel<<<(unsigned)blocks, 256, 0, stream>>>(src, dst, n);
  };

  // 0. weight / embedding conversion to bf16 (once per call; deterministic)
  cvt(img_qkv_w,  wqkv_i,  (long)HDIM * 3 * HDIM);
  cvt(txt_qkv_w,  wqkv_t,  (long)HDIM * 3 * HDIM);
  cvt(img_proj_w, wproj_i, (long)HDIM * HDIM);
  cvt(txt_proj_w, wproj_t, (long)HDIM * HDIM);
  cvt(img_mlp_w1, wmlp1_i, (long)HDIM * MLPD);
  cvt(txt_mlp_w1, wmlp1_t, (long)HDIM * MLPD);
  cvt(img_mlp_w2, wmlp2_i, (long)MLPD * HDIM);
  cvt(txt_mlp_w2, wmlp2_t, (long)MLPD * HDIM);
  cvt(ip_k_w,     wipk,    (long)HDIM * HDIM);
  cvt(ip_v_w,     wipv,    (long)HDIM * HDIM);
  cvt(image_emb,  emb16,   (long)N_IP * HDIM);

  // 1. modulation vectors
  silu_kernel<<<8, 256, 0, stream>>>(vec, sv);
  mod_kernel<<<48, 256, 0, stream>>>(sv, img_mod_w, img_mod_b, mod_img);
  mod_kernel<<<48, 256, 0, stream>>>(sv, txt_mod_w, txt_mod_b, mod_txt);

  // 2. LN + modulate (sh1 = chunk0, sc1 = chunk1) -> bf16
  lnmod_kernel<<<L_IMG, 256, 0, stream>>>(img, mod_img, mod_img + HDIM, xmod_i);
  lnmod_kernel<<<L_TXT, 256, 0, stream>>>(txt, mod_txt, mod_txt + HDIM, xmod_t);

  // 3. QKV GEMMs (f32 out for RMS precision)
  launch_gemm(stream, xmod_i, HDIM, wqkv_i, 3 * HDIM, 0, qkv_i, nullptr, 3 * HDIM,
              img_qkv_b, nullptr, nullptr, 0, L_IMG, 3 * HDIM, HDIM, 1.0f, EP_BIAS);
  launch_gemm(stream, xmod_t, HDIM, wqkv_t, 3 * HDIM, 0, qkv_t, nullptr, 3 * HDIM,
              txt_qkv_b, nullptr, nullptr, 0, L_TXT, 3 * HDIM, HDIM, 1.0f, EP_BIAS);

  // 4. RMS + RoPE + repack (txt at seq [0,256), img at [256,1280)) -> bf16
  qkvfin_kernel<<<dim3(L_TXT, NHEADS), 128, 0, stream>>>(qkv_t, L_TXT, 0,
      txt_q_scale, txt_k_scale, pe, qpk, kpk, vpk, nullptr);
  qkvfin_kernel<<<dim3(L_IMG, NHEADS), 128, 0, stream>>>(qkv_i, L_IMG, L_TXT,
      img_q_scale, img_k_scale, pe, qpk, kpk, vpk, qnope);

  // 5. IP-adapter K/V projections (M = 16, guarded)
  launch_gemm(stream, emb16, HDIM, wipk, HDIM, 0, ipk, nullptr, HDIM,
              ip_k_b, nullptr, nullptr, 0, N_IP, HDIM, HDIM, 1.0f, EP_BIAS);
  launch_gemm(stream, emb16, HDIM, wipv, HDIM, 0, ipv, nullptr, HDIM,
              ip_v_b, nullptr, nullptr, 0, N_IP, HDIM, HDIM, 1.0f, EP_BIAS);

  // 6. attention per head (S/P buffers reused; stream-ordered)
  for (int h = 0; h < NHEADS; ++h) {
    const u16* qh = qpk + (size_t)h * L_SEQ * DHEAD;
    const u16* kh = kpk + (size_t)h * L_SEQ * DHEAD;
    const u16* vh = vpk + (size_t)h * L_SEQ * DHEAD;
    launch_gemm(stream, qh, DHEAD, kh, DHEAD, 1, Sbuf, nullptr, L_SEQ,
                nullptr, nullptr, nullptr, 0, L_SEQ, L_SEQ, DHEAD, ATT_SCALE, 0);
    softmax_kernel<<<L_SEQ, 256, 0, stream>>>(Sbuf, Pbuf, L_SEQ);
    launch_gemm(stream, Pbuf, L_SEQ, vh, DHEAD, 0, attn + h * DHEAD, nullptr, HDIM,
                nullptr, nullptr, nullptr, 0, L_SEQ, DHEAD, L_SEQ, 1.0f, 0);
  }

  // 7. IP attention (adds into img rows of attn), then attn -> bf16
  ipa_kernel<<<dim3(L_IMG, NHEADS), 128, 0, stream>>>(qnope, ipk, ipv, attn);
  cvt(attn, attn16, (long)L_SEQ * HDIM);

  // 8. projections + gated residual (g1 = chunk2)
  launch_gemm(stream, attn16 + (size_t)L_TXT * HDIM, HDIM, wproj_i, HDIM, 0,
              mid_i, nullptr, HDIM, img_proj_b, mod_img + 2 * HDIM, img, HDIM,
              L_IMG, HDIM, HDIM, 1.0f, EP_BIAS | EP_RESGATE);
  launch_gemm(stream, attn16, HDIM, wproj_t, HDIM, 0,
              mid_t, nullptr, HDIM, txt_proj_b, mod_txt + 2 * HDIM, txt, HDIM,
              L_TXT, HDIM, HDIM, 1.0f, EP_BIAS | EP_RESGATE);

  // 9. second LN + modulate (sh2 = chunk3, sc2 = chunk4) -> bf16
  lnmod_kernel<<<L_IMG, 256, 0, stream>>>(mid_i, mod_img + 3 * HDIM, mod_img + 4 * HDIM, x2_i);
  lnmod_kernel<<<L_TXT, 256, 0, stream>>>(mid_t, mod_txt + 3 * HDIM, mod_txt + 4 * HDIM, x2_t);

  // 10. MLP up (bias + tanh-GELU) -> bf16 hidden
  launch_gemm(stream, x2_i, HDIM, wmlp1_i, MLPD, 0, nullptr, hid_i, MLPD,
              img_mlp_b1, nullptr, nullptr, 0, L_IMG, MLPD, HDIM, 1.0f, EP_BIAS | EP_GELU);
  launch_gemm(stream, x2_t, HDIM, wmlp1_t, MLPD, 0, nullptr, hid_t, MLPD,
              txt_mlp_b1, nullptr, nullptr, 0, L_TXT, MLPD, HDIM, 1.0f, EP_BIAS | EP_GELU);

  // 11. MLP down + gated residual (g2 = chunk5) -> final f32 outputs
  launch_gemm(stream, hid_i, MLPD, wmlp2_i, HDIM, 0, out_img, nullptr, HDIM,
              img_mlp_b2, mod_img + 5 * HDIM, mid_i, HDIM,
              L_IMG, HDIM, MLPD, 1.0f, EP_BIAS | EP_RESGATE);
  launch_gemm(stream, hid_t, MLPD, wmlp2_t, HDIM, 0, out_txt, nullptr, HDIM,
              txt_mlp_b2, mod_txt + 5 * HDIM, mid_t, HDIM,
              L_TXT, HDIM, MLPD, 1.0f, EP_BIAS | EP_RESGATE);
}
